// EgoPredictor_83107617178295
// MI455X (gfx1250) — matrix operand order repaired and checked by reading, not compiled, not run on hardware
//
#include <hip/hip_runtime.h>
#include <stdint.h>

// ---- problem constants ----
#define T_H   8
#define T_F   12
#define INSK  20
#define CAPK  32
#define DDIM  128
#define BB    256
#define NN    256
// packed rows: 2*B*CAP = 16384 ; h-rows M = 16384*8 = 131072
#define PROWS 16384
#define NEI0  8192           // first nei packed row
#define MTOT  131072
#define MHALF 65536

typedef __attribute__((ext_vector_type(16))) _Float16 v16h;
typedef __attribute__((ext_vector_type(8)))  float    v8f;
typedef __attribute__((ext_vector_type(4)))  int      i4v;
typedef __attribute__((ext_vector_type(2)))  int      i2v;

union Frag { v16h v; i4v q[2]; };
union H4   { _Float16 h[4]; i2v q; };

// closed-form least-squares fit matrices (t = 0..7, Xf rows t = 8..19)
__device__ __forceinline__ float pfitf(int h, int t) {
  return (float)h * (float)(2*t - 7) * (1.0f/84.0f) + (float)(5 - t) * (1.0f/12.0f);
}
__device__ __forceinline__ float apredf(int f, int t) {
  return (float)(8 + f) * (float)(2*t - 7) * (1.0f/84.0f) + (float)(5 - t) * (1.0f/12.0f);
}

__device__ __forceinline__ uint32_t hmix(uint32_t x) {
  x ^= x >> 16; x *= 0x7feb352du; x ^= x >> 15; x *= 0x846ca68bu; x ^= x >> 16; return x;
}
// deterministic standard normal (stands in for jax.random.normal(key(42)))
__device__ __forceinline__ float zrand(uint32_t s) {
  uint32_t a = hmix(s ^ 0x9e3779b9u);
  uint32_t b = hmix(s ^ 0x85ebca6bu);
  float u1 = ((float)((a >> 8) + 1u)) * (1.0f/16777216.0f);   // (0,1]
  float u2 = ((float)(b >> 8)) * (1.0f/16777216.0f);
  return sqrtf(-2.0f * logf(u1)) * cosf(6.28318530718f * u2);
}

// ---------------- K1: per-batch 32 nearest neighbors, index-sorted ------------
__global__ __launch_bounds__(256) void k_topk(const float* __restrict__ ego,
                                              const float* __restrict__ nei,
                                              int* __restrict__ idx) {
  __shared__ float d2[NN];
  __shared__ int   sc[NN];
  const int b = blockIdx.x, n = threadIdx.x;
  const float ex = ego[b*16 + 14], ey = ego[b*16 + 15];
  const float* p = nei + (((size_t)b*NN + n)*T_H + 7)*2;
  float dx = ex - p[0], dy = ey - p[1];
  d2[n] = dx*dx + dy*dy;
  __syncthreads();
  const float dn = d2[n];
  int rank = 0;
  for (int j = 0; j < NN; ++j) {
    float dj = d2[j];
    rank += (dj < dn) || (dj == dn && j < n);
  }
  const int flag = (rank < CAPK) ? 1 : 0;
  sc[n] = flag;
  __syncthreads();
  for (int off = 1; off < NN; off <<= 1) {          // inclusive scan, index order
    int add = (n >= off) ? sc[n - off] : 0;
    __syncthreads();
    sc[n] += add;
    __syncthreads();
  }
  if (flag) idx[b*CAPK + (sc[n] - 1)] = n;          // compaction == sort-by-index
}

// ---------------- K2: weight transpose + pad to f16 [Npad][K] -----------------
__global__ void k_wT(const float* __restrict__ W, _Float16* __restrict__ Wt,
                     int K, int N, int Npad) {
  int e = blockIdx.x*256 + threadIdx.x;             // e = n*K + k
  if (e >= Npad*K) return;
  int n = e / K, k = e - n*K;
  Wt[e] = (_Float16)((n < N) ? W[k*N + n] : 0.0f);
}

// ---------------- K3: gather + detrend + f0 + f_z (one wave per packed row) ---
__global__ __launch_bounds__(256) void k_frontend(
    const float* __restrict__ ego, const float* __restrict__ nei,
    const int* __restrict__ idx,
    _Float16* __restrict__ ACT0, _Float16* __restrict__ FZ,
    float* __restrict__ refY,
    const float* __restrict__ W_ld, const float* __restrict__ b_ld,
    const float* __restrict__ W_nz, const float* __restrict__ b_nz) {
  const int r    = blockIdx.x*8 + (threadIdx.x >> 5);   // packed row 0..16383
  const int lane = threadIdx.x & 31;
  const int isNei = (r >= NEI0);
  const float* src;
  if (!isNei) {
    src = ego + (size_t)(r >> 5) * (T_H*2);
  } else {
    int pb = r - NEI0, b = pb >> 5, c = pb & 31;
    int n = idx[b*CAPK + c];
    src = nei + ((size_t)b*NN + n) * (T_H*2);
  }
  float xc[T_H][2];
  {
    float rx = src[14], ry = src[15];
    #pragma unroll
    for (int t = 0; t < T_H; ++t) { xc[t][0] = src[t*2] - rx; xc[t][1] = src[t*2+1] - ry; }
    if (isNei) {
      int pb = r - NEI0;
      if (lane < 24) {                                 // y_linear (12x2) for nei rows
        int f = lane >> 1, d = lane & 1;
        float s = 0.f;
        #pragma unroll
        for (int t = 0; t < T_H; ++t) s += apredf(f, t) * xc[t][d];
        refY[(size_t)pb*26 + 2 + lane] = s;
      } else if (lane == 24) {
        refY[(size_t)pb*26 + 0] = rx;
        refY[(size_t)pb*26 + 1] = ry;
      }
    }
  }
  float xd[T_H][2];
  #pragma unroll
  for (int h = 0; h < T_H; ++h) {
    float l0 = 0.f, l1 = 0.f;
    #pragma unroll
    for (int t = 0; t < T_H; ++t) { float p = pfitf(h, t); l0 += p*xc[t][0]; l1 += p*xc[t][1]; }
    xd[h][0] = xc[h][0] - l0; xd[h][1] = xc[h][1] - l1;
  }
  for (int h = 0; h < T_H; ++h) {
    const size_t m = (size_t)r*T_H + h;
    const float z = zrand((uint32_t)(m*32u + (uint32_t)lane));
    H4 f0, fz;
    #pragma unroll
    for (int o = 0; o < 4; ++o) {
      const int n = lane*4 + o;
      float v = xd[h][0]*W_ld[n] + xd[h][1]*W_ld[DDIM + n] + b_ld[n];
      f0.h[o] = (_Float16)tanhf(v);
      float s = b_nz[n];
      #pragma unroll
      for (int j = 0; j < 32; ++j) s += __shfl(z, j) * W_nz[j*DDIM + n];
      fz.h[o] = (_Float16)tanhf(s);
    }
    *(i2v*)(ACT0 + m*DDIM + lane*4) = f0.q;
    *(i2v*)(FZ   + m*DDIM + lane*4) = fz.q;
  }
}

// ---------------- K4: WMMA GEMM  Out[M][Nout] = act([A0|A1] @ Wt^T + bias) ----
// Wt is f16 [NT*16][K0+K1], k-contiguous. 128 rows per block, 16 rows per wave.
// Per k-step we force the schedule "NT*2 ds_reads, then NT WMMAs" via
// sched_group_barrier so the B-fragment loads stay in distinct registers and
// the WMMAs retire behind counter-based partial dscnt waits.
template<int NT>
__global__ __launch_bounds__(256) void k_gemm(
    const _Float16* __restrict__ A0, const _Float16* __restrict__ A1,
    const _Float16* __restrict__ Wt, const float* __restrict__ bias,
    _Float16* __restrict__ Out, int K0, int K1, int Nout, int act) {
  __shared__ _Float16 smem[128*136];
  const int tid  = threadIdx.x;
  const int lane = tid & 31;
  const int wave = tid >> 5;
  const int rowBase = blockIdx.x*128 + wave*16;
  const int Ktot = K0 + K1;
  v8f acc[NT] = {};
  const int nchunks = (K1 > 0) ? 2 : 1;
  for (int ch = 0; ch < nchunks; ++ch) {
    const _Float16* Asrc = ch ? A1 : A0;
    const int Astride = ch ? K1 : K0;
    const int kcb = ch ? K0 : 0;
    __syncthreads();
    for (int e = tid; e < NT*256; e += 256) {         // stage Wt chunk (Npad x 128)
      int n = e >> 4, g = e & 15;
      *(i4v*)(smem + n*136 + g*8) = *(const i4v*)(Wt + (size_t)n*Ktot + kcb + g*8);
    }
    __syncthreads();
    // A fragment: lane = M (&15); elems 0..7 -> K = half*8+0..7, elems 8..15 -> +16
    const _Float16* arow = Asrc + (size_t)(rowBase + (lane & 15))*Astride + ((lane >> 4)*8);
    Frag a[4];
    #pragma unroll
    for (int ks = 0; ks < 4; ++ks) {
      a[ks].q[0] = *(const i4v*)(arow + ks*32);
      a[ks].q[1] = *(const i4v*)(arow + ks*32 + 16);
    }
    #pragma unroll
    for (int ks = 0; ks < 4; ++ks) {
      // B fragment: lane = N (&15); elems 0..7 -> K = half*8+0..7, elems 8..15 -> +16
      Frag bf[NT];
      #pragma unroll
      for (int nt = 0; nt < NT; ++nt) {
        const _Float16* bp = smem + (nt*16 + (lane & 15))*136 + ks*32 + ((lane >> 4)*8);
        bf[nt].q[0] = *(const i4v*)bp;
        bf[nt].q[1] = *(const i4v*)(bp + 16);
      }
      #pragma unroll
      for (int nt = 0; nt < NT; ++nt) {
        acc[nt] = __builtin_amdgcn_wmma_f32_16x16x32_f16(
            false, a[ks].v, false, bf[nt].v, (short)0, acc[nt], false, false);
      }
      // schedule: batch the 2*NT LDS reads, then the NT matrix ops
      __builtin_amdgcn_sched_group_barrier(0x0100, NT*2, 0);  // DS read group
      __builtin_amdgcn_sched_group_barrier(0x0008, NT,   0);  // WMMA group
    }
  }
  const int nloc = lane & 15, mh = (lane >> 4)*8;
  #pragma unroll
  for (int nt = 0; nt < NT; ++nt) {
    const int n = nt*16 + nloc;
    #pragma unroll
    for (int r = 0; r < 8; ++r) {
      if (n < Nout) {
        float v = acc[nt][r] + (bias ? bias[n] : 0.0f);
        if (act == 1) v = fmaxf(v, 0.0f);
        else if (act == 2) v = tanhf(v);
        Out[(size_t)(rowBase + mh + r)*Nout + n] = (_Float16)v;
      }
    }
  }
}

// ---------------- K5: default y = broadcast(A_PRED @ nei) ---------------------
__global__ __launch_bounds__(256) void k_ybase(const float* __restrict__ nei,
                                               float* __restrict__ y) {
  const int row  = blockIdx.x*8 + (threadIdx.x >> 5);   // b*256+n, 0..65535
  const int lane = threadIdx.x & 31;
  const float* p = nei + (size_t)row*(T_H*2);
  float x[T_H][2];
  #pragma unroll
  for (int t = 0; t < T_H; ++t) { x[t][0] = p[t*2]; x[t][1] = p[t*2+1]; }
  float yb[T_F][2];
  #pragma unroll
  for (int f = 0; f < T_F; ++f) {
    float s0 = 0.f, s1 = 0.f;
    #pragma unroll
    for (int t = 0; t < T_H; ++t) { float a = apredf(f, t); s0 += a*x[t][0]; s1 += a*x[t][1]; }
    yb[f][0] = s0; yb[f][1] = s1;
  }
  float* o = y + (size_t)row * (INSK*T_F*2);
  for (int j = lane; j < INSK*T_F*2; j += 32) {
    int rem = j % (T_F*2);
    o[j] = yb[rem >> 1][rem & 1];
  }
}

// ---------------- K6: fused contraction + decode + scatter --------------------
// pred[i,f,dd] = sum_h ins_k[h,i]*rev_k[h,f]*g[h,dd],  g = f_nei @ W_dec
__global__ __launch_bounds__(128) void k_final(
    const int* __restrict__ idx, const _Float16* __restrict__ Ffin,
    const _Float16* __restrict__ REVb, const _Float16* __restrict__ INSb,
    const float* __restrict__ refY, const float* __restrict__ W_dec,
    const float* __restrict__ b_dec, float* __restrict__ y) {
  __shared__ float g[16], rk[T_H*T_F], ik[T_H*INSK];
  const int pb = blockIdx.x, tid = threadIdx.x;
  const _Float16* fnei = Ffin + (size_t)(NEI0 + pb)*T_H*DDIM;
  for (int t = tid; t < 16; t += 128) {
    int h = t >> 1, dd = t & 1;
    float s = 0.f;
    for (int d = 0; d < DDIM; ++d) s += (float)fnei[h*DDIM + d] * W_dec[d*2 + dd];
    g[t] = s;
  }
  for (int t = tid; t < T_H*T_F;  t += 128) rk[t] = (float)REVb[(size_t)pb*T_H*T_F  + t];
  for (int t = tid; t < T_H*INSK; t += 128) ik[t] = (float)INSb[(size_t)pb*T_H*INSK + t];
  __syncthreads();
  const int b = pb >> 5, c = pb & 31;
  const int n = idx[b*CAPK + c];
  const float r0 = refY[(size_t)pb*26 + 0], r1 = refY[(size_t)pb*26 + 1];
  float* outrow = y + (((size_t)b*NN + n)*INSK)*T_F*2;
  for (int j = tid; j < INSK*T_F*2; j += 128) {
    int i = j / (T_F*2), rem = j % (T_F*2), f = rem >> 1, dd = rem & 1;
    float s = 0.f;
    #pragma unroll
    for (int h = 0; h < T_H; ++h) s += ik[h*INSK + i] * rk[h*T_F + f] * g[h*2 + dd];
    s += b_dec[dd] + (dd ? r1 : r0) + refY[(size_t)pb*26 + 2 + f*2 + dd];
    outrow[(size_t)i*T_F*2 + f*2 + dd] = s;
  }
}

// ---------------- host launcher ----------------------------------------------
extern "C" void kernel_launch(void* const* d_in, const int* in_sizes, int n_in,
                              void* d_out, int out_size, void* d_ws, size_t ws_size,
                              hipStream_t stream) {
  (void)in_sizes; (void)n_in; (void)out_size; (void)ws_size;
  const float* ego   = (const float*)d_in[0];
  const float* nei   = (const float*)d_in[1];
  const float* W_ld  = (const float*)d_in[2];
  const float* b_ld  = (const float*)d_in[3];
  const float* W_fc  = (const float*)d_in[4];
  const float* b_fc  = (const float*)d_in[5];
  const float* W_nz  = (const float*)d_in[6];
  const float* b_nz  = (const float*)d_in[7];
  const float* W_cc  = (const float*)d_in[8];
  const float* b_cc  = (const float*)d_in[9];
  const float* W_r1  = (const float*)d_in[10];
  const float* b_r1  = (const float*)d_in[11];
  const float* W_r2  = (const float*)d_in[12];
  const float* b_r2  = (const float*)d_in[13];
  const float* W_i1  = (const float*)d_in[14];
  const float* b_i1  = (const float*)d_in[15];
  const float* W_i2  = (const float*)d_in[16];
  const float* b_i2  = (const float*)d_in[17];
  const float* W_dec = (const float*)d_in[18];
  const float* b_dec = (const float*)d_in[19];
  float* y = (float*)d_out;

  char* w = (char*)d_ws;
  int*      IDX  = (int*)(w + 0);                               //  32 KB
  float*    REFY = (float*)(w + 32768);                         // 852 KB
  _Float16* WT   = (_Float16*)(w + 884736);                     // 340 KB
  _Float16* FZ   = (_Float16*)(w + 1224704);                    //  32 MB
  _Float16* ACT0 = (_Float16*)(w + 1224704 + 33554432ull);      //  32 MB
  _Float16* ACT1 = (_Float16*)(w + 1224704 + 2*33554432ull);    //  32 MB
  _Float16* REVb = (_Float16*)(w + 1224704 + 3*33554432ull);    // 1.5 MB
  _Float16* INSb = (_Float16*)(w + 1224704 + 3*33554432ull + 1572864ull); // 2.5 MB

  _Float16* WT_FC = WT;                 // 6 x 128x128
  _Float16* WT_CC = WT + 98304;         // 128 x 256
  _Float16* WT_R1 = WT + 131072;        // 128 x 128
  _Float16* WT_I1 = WT + 147456;        // 128 x 128
  _Float16* WT_R2 = WT + 163840;        // 16  x 128 (12 valid)
  _Float16* WT_I2 = WT + 165888;        // 32  x 128 (20 valid)

  k_topk<<<BB, 256, 0, stream>>>(ego, nei, IDX);

  for (int i = 0; i < 6; ++i)
    k_wT<<<64, 256, 0, stream>>>(W_fc + (size_t)i*DDIM*DDIM, WT_FC + (size_t)i*DDIM*DDIM,
                                 128, 128, 128);
  k_wT<<<128, 256, 0, stream>>>(W_cc, WT_CC, 256, 128, 128);
  k_wT<<< 64, 256, 0, stream>>>(W_r1, WT_R1, 128, 128, 128);
  k_wT<<< 64, 256, 0, stream>>>(W_i1, WT_I1, 128, 128, 128);
  k_wT<<<  8, 256, 0, stream>>>(W_r2, WT_R2, 128,  12,  16);
  k_wT<<< 16, 256, 0, stream>>>(W_i2, WT_I2, 128,  20,  32);

  k_frontend<<<PROWS/8, 256, 0, stream>>>(ego, nei, IDX, ACT0, FZ, REFY,
                                          W_ld, b_ld, W_nz, b_nz);

  // 6 FC layers, ping-pong ACT0<->ACT1 (relu x5, tanh last)
  for (int i = 0; i < 6; ++i) {
    const _Float16* in  = (i & 1) ? ACT1 : ACT0;
    _Float16*       out = (i & 1) ? ACT0 : ACT1;
    k_gemm<8><<<MTOT/128, 256, 0, stream>>>(in, (const _Float16*)nullptr,
        WT_FC + (size_t)i*DDIM*DDIM, b_fc + (size_t)i*DDIM, out,
        128, 0, 128, (i == 5) ? 2 : 1);
  }
  // concat layer: f_final = tanh([f | f_z] @ W_cc + b_cc)   (f is in ACT0)
  k_gemm<8><<<MTOT/128, 256, 0, stream>>>(ACT0, FZ, WT_CC, b_cc, ACT1,
                                          128, 128, 128, 2);
  // rev head (nei rows): t1 = tanh(f_nei@W_r1+b_r1); rev = t1@W_r2+b_r2
  k_gemm<8><<<MHALF/128, 256, 0, stream>>>(ACT1 + (size_t)MHALF*DDIM,
      (const _Float16*)nullptr, WT_R1, b_r1, ACT0, 128, 0, 128, 2);
  k_gemm<1><<<MHALF/128, 256, 0, stream>>>(ACT0, (const _Float16*)nullptr,
      WT_R2, b_r2, REVb, 128, 0, 12, 0);
  // ins head (ego rows)
  k_gemm<8><<<MHALF/128, 256, 0, stream>>>(ACT1, (const _Float16*)nullptr,
      WT_I1, b_i1, ACT0, 128, 0, 128, 2);
  k_gemm<2><<<MHALF/128, 256, 0, stream>>>(ACT0, (const _Float16*)nullptr,
      WT_I2, b_i2, INSb, 128, 0, 20, 0);

  k_ybase<<<(BB*NN)/8, 256, 0, stream>>>(nei, y);
  k_final<<<BB*CAPK, 128, 0, stream>>>(IDX, ACT1, REVb, INSb, REFY, W_dec, b_dec, y);
}
